// MistralAttention_11905649344875
// MI455X (gfx1250) — matrix operand order repaired and checked by reading, not compiled, not run on hardware
//
#include <hip/hip_runtime.h>
#include <hip/hip_bf16.h>
#include <math.h>

// ---------------------------------------------------------------------------
// Mistral attention block for gfx1250 (CDNA5): bf16 WMMA everywhere.
//   B=1, S=2048, H=4096, NH=32, NKV=8, HD=128, GROUPS=4
//
// Pipeline:
//   1. one-shot f32->bf16 conversion of hidden states + all weights (bandwidth
//      bound; keeps all per-use conversion VALU out of the GEMM hot loops)
//   2. QKV projections: pure-bf16 WMMA GEMMs (32x64 tile per wave)
//   3. RoPE + head-major rearrange, V transpose
//   4. flash attention, one wave per (head, 16-query tile), 16 WMMAs / 32 keys
//   5. output projection: bf16 WMMA GEMM -> f32
// ---------------------------------------------------------------------------

typedef __attribute__((ext_vector_type(16))) __bf16 v16bf;
typedef __attribute__((ext_vector_type(8)))  float  v8f;

union Frag16 {
    v16bf v;
    unsigned short u[16];
    uint4 q[2];
};

__device__ __forceinline__ unsigned short f2bf(float x) {
    unsigned int u = __float_as_uint(x);
    unsigned int r = u + 0x7FFFu + ((u >> 16) & 1u);   // round-to-nearest-even
    return (unsigned short)(r >> 16);
}
__device__ __forceinline__ float bf2f(unsigned short b) {
    return __uint_as_float(((unsigned int)b) << 16);
}
__device__ __forceinline__ v8f wmma_bf16(const v16bf a, const v16bf b, v8f c) {
    return __builtin_amdgcn_wmma_f32_16x16x32_bf16(
        /*neg_a=*/false, a, /*neg_b=*/false, b,
        /*c_mod=*/(short)0, c, /*reuse_a=*/false, /*reuse_b=*/false);
}

// ---------------------------------------------------------------------------
// One-shot f32 -> bf16 conversion (vectorized, bandwidth bound).
// ---------------------------------------------------------------------------
__global__ void conv_f32_bf16(const float* __restrict__ src,
                              unsigned short* __restrict__ dst, int n4) {
    int idx = blockIdx.x * blockDim.x + threadIdx.x;
    if (idx >= n4) return;
    float4 f = ((const float4*)src)[idx];
    ushort4 o;
    o.x = f2bf(f.x); o.y = f2bf(f.y); o.z = f2bf(f.z); o.w = f2bf(f.w);
    ((ushort4*)dst)[idx] = o;
}

// ---------------------------------------------------------------------------
// GEMM: C[m,n] = sum_k A[m,k] * B[n,k]
//   A: MxK bf16 row-major, B: NxK bf16 row-major (contraction contiguous in
//   both -> A and B fragments are pure 16B loads in the contiguous k order used here).
// 128 threads = 4 waves; each wave owns a 32x64 tile (2x4 fragments,
// 8 WMMAs per 32-wide k step); block covers 64x128.
// ---------------------------------------------------------------------------
template <bool OUT_BF16>
__device__ __forceinline__ void gemm_core(const unsigned short* __restrict__ A,
                                          const unsigned short* __restrict__ B,
                                          void* __restrict__ Cout,
                                          int N, int K) {
    const int lane = threadIdx.x & 31;
    const int wave = threadIdx.x >> 5;
    const int l16  = lane & 15;
    const int half = lane >> 4;
    const int m0 = blockIdx.y * 64  + (wave >> 1) * 32;
    const int n0 = blockIdx.x * 128 + (wave & 1) * 64;

    v8f acc[2][4];
    #pragma unroll
    for (int i = 0; i < 2; i++)
        #pragma unroll
        for (int j = 0; j < 4; j++)
            acc[i][j] = (v8f){0.f,0.f,0.f,0.f,0.f,0.f,0.f,0.f};

    for (int k = 0; k < K; k += 32) {
        Frag16 a[2], b[4];
        #pragma unroll
        for (int i = 0; i < 2; i++) {
            // A 16x32 bf16: half 0 -> K 0..15, half 1 -> 16..31
            const unsigned short* pa = A + (size_t)(m0 + 16 * i + l16) * K + k;
            a[i].q[0] = *(const uint4*)(pa + 16 * half);
            a[i].q[1] = *(const uint4*)(pa + 16 * half + 8);
        }
        #pragma unroll
        for (int j = 0; j < 4; j++) {
            // B 32x16 bf16: lane = column n; half 0 -> K 0..15, half 1 -> 16..31
            const unsigned short* pb = B + (size_t)(n0 + 16 * j + l16) * K + k + 16 * half;
            b[j].q[0] = *(const uint4*)(pb);
            b[j].q[1] = *(const uint4*)(pb + 8);
        }
        #pragma unroll
        for (int i = 0; i < 2; i++)
            #pragma unroll
            for (int j = 0; j < 4; j++)
                acc[i][j] = wmma_bf16(a[i].v, b[j].v, acc[i][j]);
    }

    // C/D 16x16 f32 layout: lanes 0-15 rows v, lanes 16-31 rows v+8, col = lane&15
    #pragma unroll
    for (int i = 0; i < 2; i++)
        #pragma unroll
        for (int j = 0; j < 4; j++)
            #pragma unroll
            for (int v = 0; v < 8; v++) {
                int row = m0 + 16 * i + v + 8 * half;
                int col = n0 + 16 * j + l16;
                if (OUT_BF16)
                    ((unsigned short*)Cout)[(size_t)row * N + col] = f2bf(acc[i][j][v]);
                else
                    ((float*)Cout)[(size_t)row * N + col] = acc[i][j][v];
            }
}

__global__ __launch_bounds__(128)
void gemm_bf16_bf16out(const unsigned short* __restrict__ A,
                       const unsigned short* __restrict__ B,
                       unsigned short* __restrict__ C, int N, int K) {
    gemm_core<true>(A, B, C, N, K);
}

__global__ __launch_bounds__(128)
void gemm_bf16_f32out(const unsigned short* __restrict__ A,
                      const unsigned short* __restrict__ B,
                      float* __restrict__ C, int N, int K) {
    gemm_core<false>(A, B, C, N, K);
}

// ---------------------------------------------------------------------------
// RoPE + rearrange [s][nh*128] -> [nh][s][128] (bf16 in, bf16 out).
// ---------------------------------------------------------------------------
__global__ void rope_arrange(const unsigned short* __restrict__ src,
                             unsigned short* __restrict__ dst, int nh) {
    int idx = blockIdx.x * blockDim.x + threadIdx.x;
    int total = 2048 * nh * 128;
    if (idx >= total) return;
    int d = idx & 127;
    int h = (idx >> 7) % nh;
    int s = idx / (nh * 128);
    const unsigned short* row = src + (size_t)s * nh * 128 + (size_t)h * 128;
    float x = bf2f(row[d]);
    int i = d & 63;
    float inv = __powf(10000.0f, -(float)(2 * i) * (1.0f / 128.0f));
    float ang = (float)s * inv;
    float c = __cosf(ang), sn = __sinf(ang);
    float partner = (d < 64) ? -bf2f(row[d + 64]) : bf2f(row[d - 64]);
    dst[(((size_t)h * 2048 + s) << 7) + d] = f2bf(x * c + partner * sn);
}

// V rearrange+transpose: [s][kv*128+d] -> [kv*128+d][s]
__global__ void v_transpose(const unsigned short* __restrict__ src,
                            unsigned short* __restrict__ dst) {
    int idx = blockIdx.x * blockDim.x + threadIdx.x;
    if (idx >= 2048 * 1024) return;
    int o = idx & 1023;
    int s = idx >> 10;
    dst[(size_t)o * 2048 + s] = src[(size_t)s * 1024 + o];
}

// ---------------------------------------------------------------------------
// Flash attention: one wave (32 threads) per (head, 16-query tile).
// Per 32-key block: 8 WMMAs for Q.K^T, online softmax, LDS transpose of P,
// 8 WMMAs for P.V into 16x128 f32 accumulators.
// ---------------------------------------------------------------------------
__global__ __launch_bounds__(32)
void attn_kernel(const unsigned short* __restrict__ Qr,   // [32][2048][128] bf16 (roped)
                 const unsigned short* __restrict__ Kr,   // [8][2048][128] bf16 (roped)
                 const unsigned short* __restrict__ Vt,   // [8][128][2048] bf16
                 unsigned short* __restrict__ Ab) {       // [2048][4096] bf16
    __shared__ __align__(16) unsigned short Pl[16 * 32];

    const int lane = threadIdx.x;
    const int l16  = lane & 15;
    const int half = lane >> 4;
    const int h  = blockIdx.x >> 7;     // head 0..31
    const int qt = blockIdx.x & 127;    // query tile
    const int q0 = qt * 16;
    const int kv = h >> 2;              // GQA: 4 query heads per kv head

    // Preload the 16x128 Q tile as 4 A-fragments (d chunks of 32).
    Frag16 aq[4];
    const unsigned short* qbase = Qr + (((size_t)h * 2048 + q0 + l16) << 7);
    #pragma unroll
    for (int j = 0; j < 4; j++) {
        aq[j].q[0] = *(const uint4*)(qbase + 32 * j + 16 * half);
        aq[j].q[1] = *(const uint4*)(qbase + 32 * j + 16 * half + 8);
    }

    v8f o[8];
    #pragma unroll
    for (int n = 0; n < 8; n++) o[n] = (v8f){0.f,0.f,0.f,0.f,0.f,0.f,0.f,0.f};
    float mx[8], lsum[8];
    #pragma unroll
    for (int v = 0; v < 8; v++) { mx[v] = -1e30f; lsum[v] = 0.0f; }

    const float scale = 0.08838834764831845f;   // 1/sqrt(128)

    for (int kb = 0; kb <= q0; kb += 32) {
        // ---- scores: S[16 x 32] = Q . K^T over d=128 ----
        v8f s0 = (v8f){0.f,0.f,0.f,0.f,0.f,0.f,0.f,0.f};
        v8f s1 = (v8f){0.f,0.f,0.f,0.f,0.f,0.f,0.f,0.f};
        #pragma unroll
        for (int j = 0; j < 4; j++) {
            Frag16 b0, b1;
            const unsigned short* kb0 =
                Kr + (((size_t)kv * 2048 + kb + l16) << 7) + 32 * j + 16 * half;
            b0.q[0] = *(const uint4*)(kb0);
            b0.q[1] = *(const uint4*)(kb0 + 8);
            const unsigned short* kb1 = kb0 + (16 << 7);
            b1.q[0] = *(const uint4*)(kb1);
            b1.q[1] = *(const uint4*)(kb1 + 8);
            s0 = wmma_bf16(aq[j].v, b0.v, s0);
            s1 = wmma_bf16(aq[j].v, b1.v, s1);
        }

        // ---- causal mask + online softmax (row stats per C-layout row) ----
        #pragma unroll
        for (int v = 0; v < 8; v++) {
            int row = q0 + v + 8 * half;
            int k0 = kb + l16;
            int k1 = kb + 16 + l16;
            float x0 = (k0 <= row) ? s0[v] * scale : -1e30f;
            float x1 = (k1 <= row) ? s1[v] * scale : -1e30f;
            float rm = fmaxf(x0, x1);
            rm = fmaxf(rm, __shfl_xor(rm, 1, 32));
            rm = fmaxf(rm, __shfl_xor(rm, 2, 32));
            rm = fmaxf(rm, __shfl_xor(rm, 4, 32));
            rm = fmaxf(rm, __shfl_xor(rm, 8, 32));
            float mnew  = fmaxf(mx[v], rm);
            float alpha = __expf(mx[v] - mnew);
            float e0 = __expf(x0 - mnew);
            float e1 = __expf(x1 - mnew);
            float rs = e0 + e1;
            rs += __shfl_xor(rs, 1, 32);
            rs += __shfl_xor(rs, 2, 32);
            rs += __shfl_xor(rs, 4, 32);
            rs += __shfl_xor(rs, 8, 32);
            lsum[v] = lsum[v] * alpha + rs;
            mx[v] = mnew;
            #pragma unroll
            for (int n = 0; n < 8; n++) o[n][v] *= alpha;
            // stage P into LDS (row-major 16x32) for C->A layout transpose
            Pl[(v + 8 * half) * 32 + l16]      = f2bf(e0);
            Pl[(v + 8 * half) * 32 + 16 + l16] = f2bf(e1);
        }
        __syncthreads();   // single-wave workgroup: compiler fence + DS order

        Frag16 ap;
        const unsigned short* prow = &Pl[l16 * 32];
        ap.q[0] = *(const uint4*)(prow + 16 * half);
        ap.q[1] = *(const uint4*)(prow + 16 * half + 8);
        __syncthreads();

        // ---- out += P (16x32) . V (32x128) ----
        #pragma unroll
        for (int n = 0; n < 8; n++) {
            Frag16 bv;
            const unsigned short* vb =
                Vt + ((size_t)kv * 128 + n * 16 + l16) * 2048 + kb + 16 * half;
            bv.q[0] = *(const uint4*)(vb);
            bv.q[1] = *(const uint4*)(vb + 8);
            o[n] = wmma_bf16(ap.v, bv.v, o[n]);
        }
    }

    // ---- finalize: divide by row sums, write [s][h*128+d] bf16 ----
    #pragma unroll
    for (int v = 0; v < 8; v++) {
        float inv = 1.0f / lsum[v];
        int row = q0 + v + 8 * half;
        #pragma unroll
        for (int n = 0; n < 8; n++)
            Ab[(size_t)row * 4096 + h * 128 + n * 16 + l16] = f2bf(o[n][v] * inv);
    }
}

// ---------------------------------------------------------------------------
// Host-side launch
// ---------------------------------------------------------------------------
extern "C" void kernel_launch(void* const* d_in, const int* in_sizes, int n_in,
                              void* d_out, int out_size, void* d_ws, size_t ws_size,
                              hipStream_t stream) {
    (void)in_sizes; (void)n_in; (void)out_size; (void)ws_size;

    const float* hs = (const float*)d_in[0];   // [2048][4096]
    const float* Wq = (const float*)d_in[1];   // [4096][4096]
    const float* Wk = (const float*)d_in[2];   // [1024][4096]
    const float* Wv = (const float*)d_in[3];   // [1024][4096]
    const float* Wo = (const float*)d_in[4];   // [4096][4096]
    float* out = (float*)d_out;                // [2048][4096]

    char* ws = (char*)d_ws;
    size_t off = 0;
    auto take = [&](size_t bytes) -> char* {
        char* p = ws + off;
        off += (bytes + 255) & ~(size_t)255;
        return p;
    };
    const size_t SH  = (size_t)2048 * 4096;   // S*H elements
    const size_t SO  = (size_t)2048 * 1024;   // S*(NKV*HD)
    const size_t WQE = (size_t)4096 * 4096;
    const size_t WKE = (size_t)1024 * 4096;

    unsigned short* Xb  = (unsigned short*)take(SH  * 2); // hidden bf16
    unsigned short* Wqb = (unsigned short*)take(WQE * 2);
    unsigned short* Wkb = (unsigned short*)take(WKE * 2);
    unsigned short* Wvb = (unsigned short*)take(WKE * 2);
    unsigned short* Wob = (unsigned short*)take(WQE * 2);
    unsigned short* Qb  = (unsigned short*)take(SH  * 2); // proj Q [s][o]
    unsigned short* Kb  = (unsigned short*)take(SO  * 2); // proj K [s][o]
    unsigned short* Vb  = (unsigned short*)take(SO  * 2); // proj V [s][o]
    unsigned short* Qr  = (unsigned short*)take(SH  * 2); // roped Q [h][s][d]
    unsigned short* Kr  = (unsigned short*)take(SO  * 2); // roped K [kv][s][d]
    unsigned short* Vt  = (unsigned short*)take(SO  * 2); // V^T [kv][d][s]
    unsigned short* Ab  = (unsigned short*)take(SH  * 2); // attn out [s][o]

    // 1. one-shot conversions (bandwidth bound)
    auto conv = [&](const float* s, unsigned short* d, size_t n) {
        int n4 = (int)(n / 4);
        conv_f32_bf16<<<(n4 + 255) / 256, 256, 0, stream>>>(s, d, n4);
    };
    conv(hs, Xb, SH);
    conv(Wq, Wqb, WQE);
    conv(Wk, Wkb, WKE);
    conv(Wv, Wvb, WKE);
    conv(Wo, Wob, WQE);

    dim3 blk(128);
    // 2. QKV projections (pure bf16 WMMA GEMMs)
    gemm_bf16_bf16out<<<dim3(4096 / 128, 2048 / 64), blk, 0, stream>>>(Xb, Wqb, Qb, 4096, 4096);
    gemm_bf16_bf16out<<<dim3(1024 / 128, 2048 / 64), blk, 0, stream>>>(Xb, Wkb, Kb, 1024, 4096);
    gemm_bf16_bf16out<<<dim3(1024 / 128, 2048 / 64), blk, 0, stream>>>(Xb, Wvb, Vb, 1024, 4096);

    // 3. RoPE + layout rearrange
    rope_arrange<<<(int)((SH + 255) / 256), 256, 0, stream>>>(Qb, Qr, 32);
    rope_arrange<<<(int)((SO + 255) / 256), 256, 0, stream>>>(Kb, Kr, 8);
    v_transpose<<<(int)((SO + 255) / 256), 256, 0, stream>>>(Vb, Vt);

    // 4. Flash attention: 32 heads x 128 query tiles, one wave each
    attn_kernel<<<32 * 128, 32, 0, stream>>>(Qr, Kr, Vt, Ab);

    // 5. Output projection -> f32
    gemm_bf16_f32out<<<dim3(4096 / 128, 2048 / 64), blk, 0, stream>>>(Ab, Wob, out, 4096, 4096);
}
